// CNNModel_11201274708408
// MI455X (gfx1250) — matrix operand order repaired and checked by reading, not compiled, not run on hardware
//
#include <hip/hip_runtime.h>

typedef _Float16 f16;
typedef __attribute__((ext_vector_type(16))) _Float16 v16h;
typedef __attribute__((ext_vector_type(8)))  float    v8f;

#define B_DIM 64
#define N_DIM 256
#define L_DIM 1024
#define TLEN  100
#define WAVES 8

__global__ void zero_out_kernel(float* __restrict__ out, int n) {
  int i = blockIdx.x * blockDim.x + threadIdx.x;
  if (i < n) out[i] = 0.f;
}

__global__ __launch_bounds__(256) void cnn_fused_kernel(
    const int* __restrict__ tokens, const int* __restrict__ lengths,
    const float* __restrict__ embed_w, const float* __restrict__ conv_w,
    const float* __restrict__ conv_b, const float* __restrict__ w1,
    const float* __restrict__ b1, const float* __restrict__ w2,
    const float* __restrict__ b2, const float* __restrict__ wc,
    const float* __restrict__ bc, float* __restrict__ out)
{
  // block-shared padded weight tiles (WMMA layouts read these)
  __shared__ f16   wconv[32 * 16];   // conv_w [K=32][N=16], N>=12 zero
  __shared__ f16   w1sh [32 * 32];   // w1     [K=32][N=32], K>=12 zero
  __shared__ float w2sh [32 * 64];   // w2     [32][64] f32
  // per-wave scratch
  __shared__ f16   xbuf[WAVES][448]; // resized [100][4] + zero pad
  __shared__ f16   zsh [WAVES][512]; // pooled A for FC1: [16][32], padded
  __shared__ float gsh [WAVES][32];  // global-max features
  __shared__ float fsh [WAVES][64];  // FC2 output

  const int tid  = threadIdx.x;
  const int wv   = tid >> 5;
  const int lane = tid & 31;
  const int half = lane >> 4;   // 0: lanes 0-15, 1: lanes 16-31
  const int mrow = lane & 15;

  // ---- stage weights into LDS (once per block) ----
  for (int idx = tid; idx < 512; idx += 256) {
    int k = idx >> 4, n = idx & 15;                 // k = w*4+i
    wconv[idx] = (f16)((n < 12) ? conv_w[k * 12 + n] : 0.f);
  }
  for (int idx = tid; idx < 1024; idx += 256) {
    int k = idx >> 5, n = idx & 31;
    w1sh[idx] = (f16)((k < 12) ? w1[k * 32 + n] : 0.f);
  }
  for (int idx = tid; idx < 2048; idx += 256) w2sh[idx] = w2[idx];

  // ---- this wave's node ----
  const int node = blockIdx.x * WAVES + wv;         // < 16384
  const int bidx = node >> 8;
  const int len  = lengths[node];
  const int* __restrict__ row = tokens + (size_t)node * L_DIM;

  // zero pad regions
  for (int j = lane; j < 48;  j += 32) xbuf[wv][400 + j] = (f16)0.f;
  for (int j = lane; j < 512; j += 32) zsh[wv][j]        = (f16)0.f;

  // ---- bilinear resize (half-pixel centers) -> xbuf[100][4] f16 ----
  const float lf   = (float)len;
  const int   lmax = len - 1;
  #pragma unroll
  for (int rpt = 0; rpt < 4; ++rpt) {
    int i = lane + 32 * rpt;
    if (i < TLEN) {
      float src  = (i + 0.5f) * lf * (1.0f / TLEN) - 0.5f;
      float fl   = floorf(src);
      float frac = src - fl;
      int lo = (int)fl;     lo = lo < 0 ? 0 : (lo > lmax ? lmax : lo);
      int hi = (int)fl + 1; hi = hi < 0 ? 0 : (hi > lmax ? lmax : hi);
      int tl = row[lo], th = row[hi];
      #pragma unroll
      for (int c = 0; c < 4; ++c) {
        float v = embed_w[tl * 4 + c] * (1.f - frac) + embed_w[th * 4 + c] * frac;
        xbuf[wv][4 * i + c] = (f16)v;
      }
    }
  }
  __syncthreads();

  // ---- conv B fragment (K=32 x N=16): K = e + (e&8) + 8*half
  v16h bconv;
  #pragma unroll
  for (int e = 0; e < 16; ++e)
    bconv[e] = wconv[(e + (e & 8) + 8 * half) * 16 + mrow];

  const float cb = conv_b[mrow < 12 ? mrow : 11];

  // ---- conv1d as im2col GEMM (6 M-tiles) + fused bias/ReLU/maxpool(8) ----
  #pragma unroll
  for (int T = 0; T < 6; ++T) {
    v16h a;
    int base = 4 * (16 * T + mrow) + 8 * half;       // im2col row = contiguous slice
    #pragma unroll
    for (int e = 0; e < 16; ++e)
      a[e] = xbuf[wv][base + e + (e & 8)];           // A layout: K = e + (e&8) + 8*half
    v8f c = {};
    v8f d = __builtin_amdgcn_wmma_f32_16x16x32_f16(false, a, false, bconv,
                                                   (short)0, c, false, false);
    // VGPR r holds output position p = 16T + 8*half + r, channel n = mrow:
    // one half-wave's 8 accumulators == one pool-of-8 window.
    float pmax = 0.f;
    #pragma unroll
    for (int r = 0; r < 8; ++r) {
      float v = d[r] + cb;
      v = v > 0.f ? v : 0.f;
      pmax = v > pmax ? v : pmax;
    }
    int t = 2 * T + half;                            // pooled index, keep t<11
    if (t < 11 && mrow < 12) zsh[wv][t * 32 + mrow] = (f16)pmax;
  }
  __syncthreads();

  // ---- FC1: [16x32(K pad)] @ [32x32] via two WMMAs, fused ReLU + max over t ----
  v16h az;
  {
    int base = mrow * 32 + 8 * half;
    #pragma unroll
    for (int e = 0; e < 16; ++e)
      az[e] = zsh[wv][base + e + (e & 8)];
  }
  #pragma unroll
  for (int j = 0; j < 2; ++j) {
    v16h bw;
    #pragma unroll
    for (int e = 0; e < 16; ++e)
      bw[e] = w1sh[(e + (e & 8) + 8 * half) * 32 + 16 * j + mrow];
    v8f c = {};
    v8f d = __builtin_amdgcn_wmma_f32_16x16x32_f16(false, az, false, bw,
                                                   (short)0, c, false, false);
    float bb   = b1[16 * j + mrow];
    float pmax = 0.f;
    int   rlim = half ? 3 : 8;                       // rows m = r + 8*half, valid m<=10
    #pragma unroll
    for (int r = 0; r < 8; ++r) {
      if (r < rlim) {
        float v = d[r] + bb;
        v = v > 0.f ? v : 0.f;
        pmax = v > pmax ? v : pmax;
      }
    }
    float other = __shfl_xor(pmax, 16, 32);
    pmax = pmax > other ? pmax : other;
    if (half == 0) gsh[wv][16 * j + mrow] = pmax;
  }
  __syncthreads();

  // ---- FC2 (32->64) + ReLU, per-lane VALU ----
  #pragma unroll
  for (int p = 0; p < 2; ++p) {
    int n = lane + 32 * p;
    float acc = b2[n];
    #pragma unroll 8
    for (int k = 0; k < 32; ++k) acc += gsh[wv][k] * w2sh[k * 64 + n];
    fsh[wv][n] = acc > 0.f ? acc : 0.f;
  }
  __syncthreads();

  // ---- classifier + softmax + mean over N via atomic ----
  if (lane == 0) {
    float s0 = bc[0], s1 = bc[1];
    for (int k = 0; k < 64; ++k) {
      float fv = fsh[wv][k];
      s0 += fv * wc[k * 2 + 0];
      s1 += fv * wc[k * 2 + 1];
    }
    float m  = s0 > s1 ? s0 : s1;
    float e0 = __expf(s0 - m), e1 = __expf(s1 - m);
    float inv = 1.f / (e0 + e1);
    atomicAdd(&out[bidx * 2 + 0], e0 * inv * (1.0f / N_DIM));
    atomicAdd(&out[bidx * 2 + 1], e1 * inv * (1.0f / N_DIM));
  }
}

extern "C" void kernel_launch(void* const* d_in, const int* in_sizes, int n_in,
                              void* d_out, int out_size, void* d_ws, size_t ws_size,
                              hipStream_t stream) {
  const int*   tokens  = (const int*)  d_in[0];
  const int*   lengths = (const int*)  d_in[1];
  const float* embed_w = (const float*)d_in[2];
  const float* conv_w  = (const float*)d_in[3];
  const float* conv_b  = (const float*)d_in[4];
  const float* w1      = (const float*)d_in[5];
  const float* b1      = (const float*)d_in[6];
  const float* w2      = (const float*)d_in[7];
  const float* b2      = (const float*)d_in[8];
  const float* wc      = (const float*)d_in[9];
  const float* bc      = (const float*)d_in[10];
  float* out = (float*)d_out;

  // out is poisoned by the harness; zero it before atomic accumulation
  zero_out_kernel<<<1, 128, 0, stream>>>(out, out_size);

  const int nodes  = B_DIM * N_DIM;          // 16384
  const int blocks = nodes / WAVES;          // 2048 blocks x 256 threads (8 waves)
  cnn_fused_kernel<<<blocks, 256, 0, stream>>>(tokens, lengths, embed_w, conv_w,
                                               conv_b, w1, b1, w2, b2, wc, bc, out);
}
